// RVT_block_29463475650622
// MI455X (gfx1250) — matrix operand order repaired and checked by reading, not compiled, not run on hardware
//
#include <hip/hip_runtime.h>
#include <hip/hip_bf16.h>

typedef __attribute__((ext_vector_type(16))) _Float16 v16h;
typedef __attribute__((ext_vector_type(8)))  _Float16 v8h;
typedef __attribute__((ext_vector_type(8)))  float    v8f;

#define DEV __device__ __forceinline__

constexpr int BB = 32, HH = 128, WWn = 128, CC = 80, DH = 10;
constexpr int CP = 96;    // C padded to multiple of 32 (K dim)
constexpr int C3 = 240;   // 3*C
constexpr int C4 = 320;   // 4*C
constexpr int NTOK = BB * HH * WWn;   // 524288 tokens
constexpr int NWIN = NTOK / 64;       // 8192 windows / token tiles

// ---------------------------------------------------------------------------
// WMMA helpers (CDNA5 v_wmma_f32_16x16x32_f16, contiguous k order, same in A and B)
// ---------------------------------------------------------------------------
DEV v8f wmma_f16(v16h a, v16h b, v8f c) {
  return __builtin_amdgcn_wmma_f32_16x16x32_f16(
      /*neg_a=*/false, a, /*neg_b=*/false, b,
      /*c_mod=*/(short)0, c, /*reuse_a=*/false, /*reuse_b=*/false);
}

// A tile 16x32 (MxK) from row-major f16 LDS [rows][ld], ld % 8 == 0.
// lane<16: row M=lane, halves 0-15 -> K 0..15
// lane>=16: row M=lane-16, halves 0-15 -> K 16..31
DEV v16h ldsA(const _Float16* p, int m0, int k0, int ld) {
  const int lane = threadIdx.x & 31;
  const _Float16* r = p + (m0 + (lane & 15)) * ld + k0 + ((lane >> 4) << 4);
  const v8h lo = *(const v8h*)r;        // 16B aligned -> ds_load_b128
  const v8h hi = *(const v8h*)(r + 8);
  return __builtin_shufflevector(lo, hi, 0, 1, 2, 3, 4, 5, 6, 7,
                                 8, 9, 10, 11, 12, 13, 14, 15);
}

// B tile 32x16 (KxN) where LDS holds the N-major (transposed) image:
// B(k,n) = p[n*ld + k], ld % 8 == 0.  Contiguous 16 halfs per lane.
// lane<16: col N=lane, K 0..15 ; lane>=16: K 16..31
DEV v16h ldsBt(const _Float16* p, int k0, int n0, int ld) {
  const int lane = threadIdx.x & 31;
  const _Float16* r = p + (n0 + (lane & 15)) * ld + k0 + ((lane >> 4) << 4);
  const v8h lo = *(const v8h*)r;
  const v8h hi = *(const v8h*)(r + 8);
  return __builtin_shufflevector(lo, hi, 0, 1, 2, 3, 4, 5, 6, 7,
                                 8, 9, 10, 11, 12, 13, 14, 15);
}

// ---------------------------------------------------------------------------
// fp32 [K][N] -> f16 transposed [N][KP] with zero K-padding (into workspace)
// ---------------------------------------------------------------------------
__global__ void convert_weights_t(const float* __restrict__ w, _Float16* __restrict__ dst,
                                  int K, int N, int KP) {
  int idx = blockIdx.x * blockDim.x + threadIdx.x;
  if (idx >= N * KP) return;
  int n = idx / KP, k = idx - n * KP;
  dst[idx] = (k < K) ? (_Float16)w[k * N + n] : (_Float16)0.f;
}

// token(t) -> pixel offset for block windows (G=false) or grid windows (G=true)
template <bool G>
DEV int pixBase(int wid, int t) {
  const int b = wid >> 8, rem = wid & 255;
  const int r = t >> 3, c = t & 7;
  int ph, pw;
  if (!G) { ph = ((rem >> 4) << 3) + r; pw = ((rem & 15) << 3) + c; }
  else    { ph = (r << 4) + (rem >> 4); pw = (c << 4) + (rem & 15); }
  return ((b * HH + ph) * WWn + pw) * CC;
}

// ---------------------------------------------------------------------------
// Fused l2norm + windowed MHSA + proj + residual. One WG per 64-token window.
// ---------------------------------------------------------------------------
template <bool G>
__global__ void __launch_bounds__(256)
attn_kernel(const float* __restrict__ xin,
            const _Float16* __restrict__ wqkv_t, const float* __restrict__ bqkv,
            const _Float16* __restrict__ wproj_t, const float* __restrict__ bproj,
            float* __restrict__ xout) {
  extern __shared__ char smem[];
  _Float16* xnh = (_Float16*)smem;          // [64][96]   normalized x, f16
  _Float16* qh  = xnh + 64 * CP;            // [64][8*32] token-major q
  _Float16* kh  = qh + 64 * 256;            // [64][8*32] token-major k
  _Float16* vt  = kh + 64 * 256;            // [8*32][64] dim-major v
  _Float16* pr  = vt + 256 * 64;            // [8][64][64] softmax probs
  _Float16* oh  = pr + 8 * 64 * 64;         // [64][96]   attn output
  _Float16* wq  = oh + 64 * CP;             // [240][96]  qkv weight, N-major
  _Float16* wp  = wq + C3 * CP;             // [80][96]   proj weight, N-major
  float* xn32   = (float*)(wp + CC * CP);   // [64][80]   normalized x, f32
  float* out32  = (float*)pr;               // reuse probs region for writeback
  __shared__ float red[64][4];

  const int tid = threadIdx.x;
  const int wid = blockIdx.x;

  // zero padded regions (head-dim pad 10->32, K pad 80->96), 128b stores
  {
    const v8h z8 = {};
    for (int i = tid; i < 64 * 256 / 8; i += 256) {
      ((v8h*)qh)[i] = z8; ((v8h*)kh)[i] = z8; ((v8h*)vt)[i] = z8;
    }
    for (int i = tid; i < 64 * CP / 8; i += 256) {
      ((v8h*)xnh)[i] = z8; ((v8h*)oh)[i] = z8;
    }
  }

  // load 64 tokens x 80 ch, l2-normalize (4 threads per token)
  const int t = tid >> 2, sub = tid & 3;
  const int base = pixBase<G>(wid, t) + sub * 20;
  float v[20];
  float ss = 0.f;
#pragma unroll
  for (int i = 0; i < 20; ++i) { v[i] = xin[base + i]; ss += v[i] * v[i]; }
  red[t][sub] = ss;
  __syncthreads();
  const float inv = 1.f / fmaxf(sqrtf(red[t][0] + red[t][1] + red[t][2] + red[t][3]), 1e-12f);
#pragma unroll
  for (int i = 0; i < 20; ++i) {
    const float xv = v[i] * inv;
    xn32[t * CC + sub * 20 + i] = xv;
    xnh[t * CP + sub * 20 + i] = (_Float16)xv;
  }
  // stage weights as 128-bit copies (L2-resident after first WG)
  for (int i = tid; i < C3 * CP / 8; i += 256) ((v8h*)wq)[i] = ((const v8h*)wqkv_t)[i];
  for (int i = tid; i < CC * CP / 8; i += 256) ((v8h*)wp)[i] = ((const v8h*)wproj_t)[i];
  __syncthreads();

  const int lane = tid & 31;
  const int wav = tid >> 5;
  const int rb = (lane >> 4) << 3;  // C-tile row base: 0 or 8
  const int cb = lane & 15;         // C-tile column

  // --- QKV GEMM: [64x96] @ [96x240], 60 tiles over 8 waves ----------------
  for (int tile = wav; tile < 60; tile += 8) {
    const int mt = tile & 3, nt = tile >> 2;
    v8f acc = {};
#pragma unroll
    for (int kt = 0; kt < 3; ++kt)
      acc = wmma_f16(ldsA(xnh, mt * 16, kt * 32, CP), ldsBt(wq, kt * 32, nt * 16, CP), acc);
    const int col = nt * 16 + cb;            // 0..239
    const int head = col / 30, rm = col % 30;
    const int which = rm / 10, d = rm % 10;  // q/k/v, dim within head
    const float bias = bqkv[col];
    if (which == 2) {                        // v: dim-major for P@V B operand
#pragma unroll
      for (int r = 0; r < 8; ++r)
        vt[(head * 32 + d) * 64 + mt * 16 + rb + r] = (_Float16)(acc[r] + bias);
    } else {
      _Float16* dst = which ? kh : qh;
#pragma unroll
      for (int r = 0; r < 8; ++r)
        dst[(mt * 16 + rb + r) * 256 + head * 32 + d] = (_Float16)(acc[r] + bias);
    }
  }
  __syncthreads();

  // --- attention: one head per wave (8 waves == 8 heads) ------------------
  {
    const int h = wav;
    _Float16* ph = pr + h * 64 * 64;
    const float scale = 0.316227766016838f;  // 10^-0.5
    for (int mt = 0; mt < 4; ++mt) {
      const v16h aq = ldsA(qh, mt * 16, h * 32, 256);
      v8f s[4];
#pragma unroll
      for (int nt = 0; nt < 4; ++nt) {
        v8f z = {};
        s[nt] = wmma_f16(aq, ldsBt(kh, h * 32, nt * 16, 256), z);
      }
      float mrow[8];
#pragma unroll
      for (int e = 0; e < 8; ++e) mrow[e] = -3.0e38f;
#pragma unroll
      for (int nt = 0; nt < 4; ++nt)
#pragma unroll
        for (int e = 0; e < 8; ++e) { s[nt][e] *= scale; mrow[e] = fmaxf(mrow[e], s[nt][e]); }
#pragma unroll
      for (int dl = 1; dl < 16; dl <<= 1)
#pragma unroll
        for (int e = 0; e < 8; ++e) mrow[e] = fmaxf(mrow[e], __shfl_xor(mrow[e], dl, 16));
      float srow[8];
#pragma unroll
      for (int e = 0; e < 8; ++e) srow[e] = 0.f;
#pragma unroll
      for (int nt = 0; nt < 4; ++nt)
#pragma unroll
        for (int e = 0; e < 8; ++e) {
          const float ev = __expf(s[nt][e] - mrow[e]);
          s[nt][e] = ev;
          srow[e] += ev;
        }
#pragma unroll
      for (int dl = 1; dl < 16; dl <<= 1)
#pragma unroll
        for (int e = 0; e < 8; ++e) srow[e] += __shfl_xor(srow[e], dl, 16);
#pragma unroll
      for (int nt = 0; nt < 4; ++nt)
#pragma unroll
        for (int e = 0; e < 8; ++e)
          ph[(mt * 16 + rb + e) * 64 + nt * 16 + cb] = (_Float16)(s[nt][e] / srow[e]);
    }
    // O = P[64x64] @ V[64x(10 pad 16)], V held dim-major (contiguous B)
    const _Float16* vv = vt + (h * 32) * 64;
    for (int mt = 0; mt < 4; ++mt) {
      v8f o = {};
#pragma unroll
      for (int kt = 0; kt < 2; ++kt)
        o = wmma_f16(ldsA(ph, mt * 16, kt * 32, 64), ldsBt(vv, kt * 32, 0, 64), o);
      if (cb < DH) {
#pragma unroll
        for (int r = 0; r < 8; ++r)
          oh[(mt * 16 + rb + r) * CP + h * DH + cb] = (_Float16)o[r];
      }
    }
  }
  __syncthreads();   // also: all reads of pr done before out32 overwrites it

  // --- proj [64x96]@[96x80] + residual -> LDS ------------------------------
  for (int tile = wav; tile < 20; tile += 8) {
    const int mt = tile & 3, nt = tile >> 2;
    v8f acc = {};
#pragma unroll
    for (int kt = 0; kt < 3; ++kt)
      acc = wmma_f16(ldsA(oh, mt * 16, kt * 32, CP), ldsBt(wp, kt * 32, nt * 16, CP), acc);
    const int col = nt * 16 + cb;  // 0..79
    const float bias = bproj[col];
#pragma unroll
    for (int r = 0; r < 8; ++r) {
      const int row = mt * 16 + rb + r;
      out32[row * CC + col] = xn32[row * CC + col] + acc[r] + bias;
    }
  }
  __syncthreads();

  // coalesced writeback, same pattern as the load
  {
    const float* src = out32 + t * CC + sub * 20;
#pragma unroll
    for (int i = 0; i < 20; ++i) xout[base + i] = src[i];
  }
}

// ---------------------------------------------------------------------------
// Fused l2norm + MLP (80->320 relu ->80) + gamma residual. 64 tokens per WG.
// ---------------------------------------------------------------------------
template <bool RELU2>
__global__ void __launch_bounds__(256)
mlp_kernel(const float* __restrict__ xin,
           const _Float16* __restrict__ w1_t, const _Float16* __restrict__ w2_t,
           const float* __restrict__ b2, const float* __restrict__ gamma,
           float* __restrict__ xout) {
  extern __shared__ char smem[];
  _Float16* xnh = (_Float16*)smem;         // [64][96]
  _Float16* h1  = xnh + 64 * CP;           // [64][320]
  _Float16* w1  = h1 + 64 * C4;            // [320][96]  N-major
  _Float16* w2  = w1 + C4 * CP;            // [80][320]  N-major
  float* xn32   = (float*)(w2 + CC * C4);  // [64][80]
  float* out32  = (float*)w1;              // reuse w1 region after GEMM1
  __shared__ float red[64][4];

  const int tid = threadIdx.x;
  const long long tok0 = (long long)blockIdx.x * 64;

  {
    const v8h z8 = {};
    for (int i = tid; i < 64 * CP / 8; i += 256) ((v8h*)xnh)[i] = z8;
  }

  const int t = tid >> 2, sub = tid & 3;
  const long long base = (tok0 + t) * CC + sub * 20;
  float v[20];
  float ss = 0.f;
#pragma unroll
  for (int i = 0; i < 20; ++i) { v[i] = xin[base + i]; ss += v[i] * v[i]; }
  red[t][sub] = ss;
  __syncthreads();
  const float inv = 1.f / fmaxf(sqrtf(red[t][0] + red[t][1] + red[t][2] + red[t][3]), 1e-12f);
#pragma unroll
  for (int i = 0; i < 20; ++i) {
    const float xv = v[i] * inv;
    xn32[t * CC + sub * 20 + i] = xv;
    xnh[t * CP + sub * 20 + i] = (_Float16)xv;
  }
  for (int i = tid; i < C4 * CP / 8; i += 256) ((v8h*)w1)[i] = ((const v8h*)w1_t)[i];
  for (int i = tid; i < CC * C4 / 8; i += 256) ((v8h*)w2)[i] = ((const v8h*)w2_t)[i];
  __syncthreads();

  const int lane = tid & 31, wav = tid >> 5;
  const int rb = (lane >> 4) << 3, cb = lane & 15;

  // h1 = relu(xn @ w1) : [64x96]@[96x320], 80 tiles over 8 waves
  for (int tile = wav; tile < 80; tile += 8) {
    const int mt = tile & 3, nt = tile >> 2;
    v8f acc = {};
#pragma unroll
    for (int kt = 0; kt < 3; ++kt)
      acc = wmma_f16(ldsA(xnh, mt * 16, kt * 32, CP), ldsBt(w1, kt * 32, nt * 16, CP), acc);
    const int col = nt * 16 + cb;
#pragma unroll
    for (int r = 0; r < 8; ++r)
      h1[(mt * 16 + rb + r) * C4 + col] = (_Float16)fmaxf(acc[r], 0.f);
  }
  __syncthreads();   // h1 ready; w1 region now reusable as out32

  // out = xn + gamma * act(h1 @ w2 + b2) : [64x320]@[320x80] -> LDS
  for (int tile = wav; tile < 20; tile += 8) {
    const int mt = tile & 3, nt = tile >> 2;
    v8f acc = {};
#pragma unroll
    for (int kt = 0; kt < 10; ++kt)
      acc = wmma_f16(ldsA(h1, mt * 16, kt * 32, C4), ldsBt(w2, kt * 32, nt * 16, C4), acc);
    const int col = nt * 16 + cb;
    const float bias = b2[col], g = gamma[col];
#pragma unroll
    for (int r = 0; r < 8; ++r) {
      const int row = mt * 16 + rb + r;
      float hv = acc[r] + bias;
      if (RELU2) hv = fmaxf(hv, 0.f);
      out32[row * CC + col] = xn32[row * CC + col] + hv * g;
    }
  }
  __syncthreads();

  // coalesced writeback
  {
    const float* src = out32 + t * CC + sub * 20;
#pragma unroll
    for (int i = 0; i < 20; ++i) xout[base + i] = src[i];
  }
}

// ---------------------------------------------------------------------------
extern "C" void kernel_launch(void* const* d_in, const int* in_sizes, int n_in,
                              void* d_out, int out_size, void* d_ws, size_t ws_size,
                              hipStream_t stream) {
  const float* x       = (const float*)d_in[0];
  const float* bw_qkv  = (const float*)d_in[1];
  const float* bb_qkv  = (const float*)d_in[2];
  const float* bw_proj = (const float*)d_in[3];
  const float* bb_proj = (const float*)d_in[4];
  const float* b_gamma = (const float*)d_in[5];
  const float* bw_mlp1 = (const float*)d_in[6];
  const float* bw_mlp2 = (const float*)d_in[7];
  const float* bb_mlp2 = (const float*)d_in[8];
  const float* gw_qkv  = (const float*)d_in[9];
  const float* gb_qkv  = (const float*)d_in[10];
  const float* gw_proj = (const float*)d_in[11];
  const float* gb_proj = (const float*)d_in[12];
  const float* g_gamma = (const float*)d_in[13];
  const float* gw_mlp1 = (const float*)d_in[14];
  const float* gw_mlp2 = (const float*)d_in[15];
  const float* gb_mlp2 = (const float*)d_in[16];

  char* ws = (char*)d_ws;
  _Float16* bqkv_h  = (_Float16*)ws;       // [240][96]
  _Float16* bproj_h = bqkv_h + C3 * CP;    // [80][96]
  _Float16* bw1_h   = bproj_h + CC * CP;   // [320][96]
  _Float16* bw2_h   = bw1_h + C4 * CP;     // [80][320]
  _Float16* gqkv_h  = bw2_h + CC * C4;
  _Float16* gproj_h = gqkv_h + C3 * CP;
  _Float16* gw1_h   = gproj_h + CC * CP;
  _Float16* gw2_h   = gw1_h + C4 * CP;
  size_t woff = (size_t)(C3 * CP + CC * CP + C4 * CP + CC * C4) * 2 * sizeof(_Float16);
  woff = (woff + 255) & ~(size_t)255;
  float* xA = (float*)(ws + woff);
  float* xB = xA + (size_t)NTOK * CC;

  auto conv = [&](const float* src, _Float16* dst, int K, int N, int KP) {
    const int total = N * KP;
    convert_weights_t<<<(total + 255) / 256, 256, 0, stream>>>(src, dst, K, N, KP);
  };
  conv(bw_qkv, bqkv_h, CC, C3, CP);
  conv(bw_proj, bproj_h, CC, CC, CP);
  conv(bw_mlp1, bw1_h, CC, C4, CP);
  conv(bw_mlp2, bw2_h, C4, CC, C4);
  conv(gw_qkv, gqkv_h, CC, C3, CP);
  conv(gw_proj, gproj_h, CC, CC, CP);
  conv(gw_mlp1, gw1_h, CC, C4, CP);
  conv(gw_mlp2, gw2_h, C4, CC, C4);

  const size_t attnLds =
      (size_t)(64 * CP + 3 * 64 * 256 + 8 * 64 * 64 + 64 * CP + C3 * CP + CC * CP) * 2 +
      (size_t)64 * CC * 4;  // ~270 KB (within 320 KB WGP LDS)
  const size_t mlpLds =
      (size_t)(64 * CP + 64 * C4 + C4 * CP + CC * C4) * 2 + (size_t)64 * CC * 4;  // ~186 KB

  attn_kernel<false><<<NWIN, 256, attnLds, stream>>>(x, bqkv_h, bb_qkv, bproj_h, bb_proj, xA);
  mlp_kernel<true><<<NWIN, 256, mlpLds, stream>>>(xA, bw1_h, bw2_h, bb_mlp2, b_gamma, xB);
  attn_kernel<true><<<NWIN, 256, attnLds, stream>>>(xB, gqkv_h, gb_qkv, gproj_h, gb_proj, xA);
  mlp_kernel<false><<<NWIN, 256, mlpLds, stream>>>(xA, gw1_h, gw2_h, gb_mlp2, g_gamma,
                                                   (float*)d_out);
  (void)in_sizes; (void)n_in; (void)out_size; (void)ws_size;
}
